// GNNModel_60902636257604
// MI455X (gfx1250) — matrix operand order repaired and checked by reading, not compiled, not run on hardware
//
#include <hip/hip_runtime.h>

// GNN (GCN encoders + 9 residual message-passing layers + decoder) for MI455X.
// Dominant cost: [131072,256]x[256,256] MLP GEMMs -> bf16 WMMA, f32 accumulate.
// Irregular scatter (degree norm / message aggregation) -> global_atomic_add_f32.

typedef __attribute__((ext_vector_type(16))) __bf16 v16bf;
typedef __attribute__((ext_vector_type(8)))  float  v8f;

#define LDA 264   // padded LDS row stride (bf16 elems) -> 132 dwords, kills bank conflicts
#define HD  256
#define EPSLN 1e-5f

// ---------------- WMMA fragment helpers ----------------
// A (16x32 bf16, ISA 7.12.2): lanes 0-15 M=0..15; VGPR v<4 -> K=(v&3)*2 (+half*8),
// v>=4 -> K=16+(v&3)*2 (+half*8); each VGPR holds K,K+1 (contiguous -> b32 LDS read).
__device__ __forceinline__ v16bf load_a_frag(const __bf16* s, int row16, int kc, int lane) {
  const __bf16* row = s + (size_t)(row16 + (lane & 15)) * LDA;
  const int half = (lane >> 4) * 8;
  v16bf a;
#pragma unroll
  for (int v = 0; v < 8; ++v) {
    int kk = kc * 32 + ((v < 4) ? 0 : 16) + (v & 3) * 2 + half;
    a[2 * v]     = row[kk];
    a[2 * v + 1] = row[kk + 1];
  }
  return a;
}

// B fragments are pre-packed so each lane reads 16 contiguous bf16 (32 bytes).
__device__ __forceinline__ v16bf load_b_frag(const __bf16* wp, int nt, int kc, int lane) {
  return *(const v16bf*)(wp + (size_t)(((nt * 8 + kc) * 32) + lane) * 16);
}

// ---------------- weight pre-pack: fp32 [256,256] -> bf16 WMMA-B fragment order ----
// Packed index: ((nt*8+kc)*32+lane)*16+i  holds  W[K][n],
//   K = kc*32 + (i&8)*2 + (lane>>4)*8 + (i&7) ,  n = nt*16 + (lane&15).
__global__ void __launch_bounds__(256) k_pack_w(const float* __restrict__ Wf,
                                                __bf16* __restrict__ Wp) {
  int id   = blockIdx.x * 256 + threadIdx.x;     // 65536 total
  int i    = id & 15;
  int lane = (id >> 4) & 31;
  int kc   = (id >> 9) & 7;
  int nt   = id >> 12;
  int k = kc * 32 + ((i & 8) << 1) + ((lane >> 4) << 3) + (i & 7);
  int n = (nt << 4) + (lane & 15);
  Wp[id] = (__bf16)Wf[k * HD + n];
}

// ---------------- plain GEMM: Y[R,256] = X[R,256] @ W (packed bf16) --------------
__global__ void __launch_bounds__(256) k_gemm256(const float* __restrict__ X,
                                                 const __bf16* __restrict__ Wp,
                                                 float* __restrict__ Y) {
  __shared__ __bf16 sA[32 * LDA];
  const int tid = threadIdx.x, lane = tid & 31, wv = tid >> 5;
  const int rowgrp = wv >> 2, colgrp = wv & 3;
  const size_t row0 = (size_t)blockIdx.x * 32;

  for (int i = tid; i < 32 * HD; i += 256) {
    int r = i >> 8, c = i & 255;
    sA[r * LDA + c] = (__bf16)X[(row0 + r) * HD + c];
  }
  __syncthreads();

  v8f acc[4] = {};
#pragma unroll
  for (int kc = 0; kc < 8; ++kc) {
    v16bf a = load_a_frag(sA, rowgrp * 16, kc, lane);
    if (kc < 7)
      __builtin_prefetch(Wp + (size_t)(((colgrp * 4) * 8 + kc + 1) * 32 + lane) * 16, 0, 1);
#pragma unroll
    for (int t = 0; t < 4; ++t) {
      v16bf b = load_b_frag(Wp, colgrp * 4 + t, kc, lane);
      acc[t] = __builtin_amdgcn_wmma_f32_16x16x32_bf16(false, a, false, b,
                                                       (short)0, acc[t], false, false);
    }
  }
  const int m = lane & 15, hi = (lane >> 4) * 8;
#pragma unroll
  for (int t = 0; t < 4; ++t) {
    int col = colgrp * 64 + t * 16 + m;
#pragma unroll
    for (int r = 0; r < 8; ++r)
      Y[(row0 + rowgrp * 16 + r + hi) * HD + col] = acc[t][r];
  }
}

// ---------------- fused MLP: relu(X@W1+b1)@W2+b2 -> LayerNorm(g,be) [+res] -------
__global__ void __launch_bounds__(256) k_mlp(const float* __restrict__ X,
                                             const __bf16* __restrict__ Wp1,
                                             const float* __restrict__ B1,
                                             const __bf16* __restrict__ Wp2,
                                             const float* __restrict__ B2,
                                             const float* __restrict__ G,
                                             const float* __restrict__ BE,
                                             const float* __restrict__ RES,
                                             float* __restrict__ Y) {
  __shared__ __bf16 sA[32 * LDA];
  __shared__ __bf16 sH[32 * LDA];
  __shared__ float  sSum[4][32];
  __shared__ float  sSq[4][32];

  const int tid = threadIdx.x, lane = tid & 31, wv = tid >> 5;
  const int rowgrp = wv >> 2, colgrp = wv & 3;
  const int m = lane & 15, hi = (lane >> 4) * 8;
  const size_t row0 = (size_t)blockIdx.x * 32;
  const v8f vzero = {0.f, 0.f, 0.f, 0.f, 0.f, 0.f, 0.f, 0.f};

  for (int i = tid; i < 32 * HD; i += 256) {
    int r = i >> 8, c = i & 255;
    sA[r * LDA + c] = (__bf16)X[(row0 + r) * HD + c];
  }
  __syncthreads();

  // ---- fc1 ----
  v8f acc[4];
#pragma unroll
  for (int t = 0; t < 4; ++t) acc[t] = vzero;
#pragma unroll
  for (int kc = 0; kc < 8; ++kc) {
    v16bf a = load_a_frag(sA, rowgrp * 16, kc, lane);
#pragma unroll
    for (int t = 0; t < 4; ++t) {
      v16bf b = load_b_frag(Wp1, colgrp * 4 + t, kc, lane);
      acc[t] = __builtin_amdgcn_wmma_f32_16x16x32_bf16(false, a, false, b,
                                                       (short)0, acc[t], false, false);
    }
  }
  // bias + relu -> sH (bf16)
#pragma unroll
  for (int t = 0; t < 4; ++t) {
    int col = colgrp * 64 + t * 16 + m;
    float bb = B1[col];
#pragma unroll
    for (int r = 0; r < 8; ++r) {
      float v = fmaxf(acc[t][r] + bb, 0.f);
      sH[(rowgrp * 16 + r + hi) * LDA + col] = (__bf16)v;
    }
  }
  __syncthreads();

  // ---- fc2 ----
#pragma unroll
  for (int t = 0; t < 4; ++t) acc[t] = vzero;
#pragma unroll
  for (int kc = 0; kc < 8; ++kc) {
    v16bf a = load_a_frag(sH, rowgrp * 16, kc, lane);
#pragma unroll
    for (int t = 0; t < 4; ++t) {
      v16bf b = load_b_frag(Wp2, colgrp * 4 + t, kc, lane);
      acc[t] = __builtin_amdgcn_wmma_f32_16x16x32_bf16(false, a, false, b,
                                                       (short)0, acc[t], false, false);
    }
  }
#pragma unroll
  for (int t = 0; t < 4; ++t) {
    float bb = B2[colgrp * 64 + t * 16 + m];
#pragma unroll
    for (int r = 0; r < 8; ++r) acc[t][r] += bb;
  }

  // ---- LayerNorm partials (per-row sum / sumsq over this wave's 64 cols) ----
#pragma unroll
  for (int r = 0; r < 8; ++r) {
    float s = 0.f, q = 0.f;
#pragma unroll
    for (int t = 0; t < 4; ++t) { float v = acc[t][r]; s += v; q += v * v; }
#pragma unroll
    for (int msk = 1; msk < 16; msk <<= 1) {
      s += __shfl_xor(s, msk, 32);
      q += __shfl_xor(q, msk, 32);
    }
    if ((lane & 15) == 0) {
      int row = rowgrp * 16 + r + hi;
      sSum[colgrp][row] = s;
      sSq[colgrp][row]  = q;
    }
  }
  __syncthreads();

  // ---- normalize + affine (+residual), write out ----
#pragma unroll
  for (int r = 0; r < 8; ++r) {
    int rl = rowgrp * 16 + r + hi;
    float S = sSum[0][rl] + sSum[1][rl] + sSum[2][rl] + sSum[3][rl];
    float Q = sSq[0][rl] + sSq[1][rl] + sSq[2][rl] + sSq[3][rl];
    float mu  = S * (1.f / 256.f);
    float var = Q * (1.f / 256.f) - mu * mu;
    float inv = rsqrtf(var + EPSLN);
    size_t grow = row0 + rl;
#pragma unroll
    for (int t = 0; t < 4; ++t) {
      int col = colgrp * 64 + t * 16 + m;
      float v = (acc[t][r] - mu) * inv * G[col] + BE[col];
      if (RES) v += RES[grow * HD + col];
      Y[grow * HD + col] = v;
    }
  }
}

// ---------------- scalar / scatter kernels ----------------
__global__ void k_fill(float* p, float v, int n) {
  int i = blockIdx.x * 256 + threadIdx.x;
  if (i < n) p[i] = v;
}
__global__ void k_deg(float* deg, const int* __restrict__ dst, int E) {
  int e = blockIdx.x * 256 + threadIdx.x;
  if (e < E) atomicAdd(&deg[dst[e]], 1.0f);
}
__global__ void k_rsqrt(float* p, int n) {
  int i = blockIdx.x * 256 + threadIdx.x;
  if (i < n) p[i] = rsqrtf(p[i]);
}
// Y[R,FO] = X[R,FI] @ W[FI,FO]   (small FI or small FO)
__global__ void k_small_gemm(const float* __restrict__ X, const float* __restrict__ W,
                             float* __restrict__ Y, int R, int FI, int FO) {
  int idx = blockIdx.x * 256 + threadIdx.x;
  if (idx >= R * FO) return;
  int r = idx / FO, c = idx - r * FO;
  float s = 0.f;
  for (int k = 0; k < FI; ++k) s += X[r * FI + k] * W[k * FO + c];
  Y[idx] = s;
}
// AGG[dst] += dinv[src]*dinv[dst] * H[src]   (per edge, per channel)
__global__ void k_gcn_scatter(const float* __restrict__ H, float* __restrict__ AGG,
                              const int* __restrict__ src, const int* __restrict__ dst,
                              const float* __restrict__ dinv, int E, int C) {
  int idx = blockIdx.x * 256 + threadIdx.x;
  if (idx >= E * C) return;
  int e = idx / C, c = idx - e * C;
  int s = src[e], d = dst[e];
  float nrm = dinv[s] * dinv[d];
  atomicAdd(&AGG[d * C + c], nrm * H[s * C + c]);
}
// AGG += dinv^2 * H + bias  (+optional relu)
__global__ void k_selfbias(float* __restrict__ AGG, const float* __restrict__ H,
                           const float* __restrict__ dinv, const float* __restrict__ b,
                           int R, int C, int relu) {
  int idx = blockIdx.x * 256 + threadIdx.x;
  if (idx >= R * C) return;
  int r = idx / C, c = idx - r * C;
  float di = dinv[r];
  float v = AGG[idx] + di * di * H[idx] + b[c];
  if (relu) v = fmaxf(v, 0.f);
  AGG[idx] = v;
}
// AGG[dst] += HX[src] + EA[e]   (processor message aggregation, C=256)
__global__ void k_msg_scatter(const float* __restrict__ HX, const float* __restrict__ EA,
                              float* __restrict__ AGG, const int* __restrict__ src,
                              const int* __restrict__ dst, int E) {
  int idx = blockIdx.x * 256 + threadIdx.x;
  if (idx >= E * HD) return;
  int e = idx >> 8, c = idx & 255;
  atomicAdd(&AGG[dst[e] * HD + c], HX[src[e] * HD + c] + EA[idx]);
}
// out = initial[:, :3] + concat(D, initial) @ Wfc + bfc
__global__ void k_final(const float* __restrict__ D, const float* __restrict__ S0,
                        const float* __restrict__ Wfc, const float* __restrict__ Bfc,
                        float* __restrict__ out, int R) {
  int idx = blockIdx.x * 256 + threadIdx.x;
  if (idx >= R * 3) return;
  int r = idx / 3, c = idx - r * 3;
  float s = Bfc[c];
  for (int k = 0; k < 3; ++k) s += D[r * 3 + k] * Wfc[k * 3 + c];
  for (int k = 0; k < 8; ++k) s += S0[r * 8 + k] * Wfc[(3 + k) * 3 + c];
  out[idx] = S0[r * 8 + c] + s;
}

// ---------------- host orchestration ----------------
// Input flattening assumed = setup_inputs() dict insertion order:
//  0:x  1:edge_index(int32,[2,E])  2:edge_attr  3:initial_state
//  4: enc_node.c1.W  5:.b  6: enc_node.c2.W  7:.b
//  8: enc_edge.c1.W  9:.b 10: enc_edge.c2.W 11:.b
//  12+12L: proc[L].node.{fc1.W,fc1.b,fc2.W,fc2.b,g,be}, .edge.{fc1.W,fc1.b,fc2.W,fc2.b,g,be}
//  120: dec.c1.W 121:.b 122: dec.c2.W 123:.b 124: dec.fc.W 125:.b
extern "C" void kernel_launch(void* const* d_in, const int* in_sizes, int n_in,
                              void* d_out, int out_size, void* d_ws, size_t ws_size,
                              hipStream_t stream) {
  const int Nn = 8192, Ee = 131072;
  auto F = [&](int i) { return (const float*)d_in[i]; };
  const float* x      = F(0);
  const int*   ei     = (const int*)d_in[1];
  const float* eattr  = F(2);
  const float* istate = F(3);
  const int* src = ei;
  const int* dst = ei + Ee;

  // workspace carve-out
  char* w = (char*)d_ws;
  size_t off = 0;
  auto take = [&](size_t b) -> char* {
    char* p = w + off;
    off = (off + b + 255) & ~(size_t)255;
    return p;
  };
  __bf16* wpk  = (__bf16*)take((size_t)39 * 65536 * 2);
  float* dinvN = (float*)take((size_t)Nn * 4);
  float* dinvE = (float*)take((size_t)Ee * 4);
  float* bufN1 = (float*)take((size_t)Nn * HD * 4);   // hx
  float* bufN2 = (float*)take((size_t)Nn * HD * 4);
  float* bufN3 = (float*)take((size_t)Nn * HD * 4);
  float* h3    = (float*)take((size_t)Nn * 3 * 4);
  float* agg3  = (float*)take((size_t)Nn * 3 * 4);
  float* bufE1 = (float*)take((size_t)Ee * HD * 4);   // he
  float* bufE2 = (float*)take((size_t)Ee * HD * 4);   // scratch / ea
  (void)ws_size; (void)n_in; (void)in_sizes; (void)out_size;

  auto wp = [&](int slot) { return wpk + (size_t)slot * 65536; };

  // ---- pack all 256x256 weights into WMMA-B fragment order (bf16) ----
  int packsrc[39];
  packsrc[0] = 6;    // enc_node.c2.W
  packsrc[1] = 10;   // enc_edge.c2.W
  packsrc[2] = 120;  // dec.c1.W
  for (int L = 0; L < 9; ++L) {
    int b = 12 + 12 * L;
    packsrc[3 + 4 * L + 0] = b + 0;  // node.fc1.W
    packsrc[3 + 4 * L + 1] = b + 2;  // node.fc2.W
    packsrc[3 + 4 * L + 2] = b + 6;  // edge.fc1.W
    packsrc[3 + 4 * L + 3] = b + 8;  // edge.fc2.W
  }
  for (int s = 0; s < 39; ++s)
    k_pack_w<<<256, 256, 0, stream>>>(F(packsrc[s]), wp(s));

  // ---- degree normalization (deg = 1 + in-degree; dinv = rsqrt(deg)) ----
  k_fill<<<(Nn + 255) / 256, 256, 0, stream>>>(dinvN, 1.0f, Nn);
  k_deg<<<(Ee + 255) / 256, 256, 0, stream>>>(dinvN, dst, Ee);
  k_rsqrt<<<(Nn + 255) / 256, 256, 0, stream>>>(dinvN, Nn);
  k_fill<<<(Ee + 255) / 256, 256, 0, stream>>>(dinvE, 1.0f, Ee);
  k_deg<<<(Ee + 255) / 256, 256, 0, stream>>>(dinvE, dst, Ee);
  k_rsqrt<<<(Ee + 255) / 256, 256, 0, stream>>>(dinvE, Ee);

  const int gN = (Nn * HD + 255) / 256;     // elementwise over [N,256]
  const int gE = (Ee * HD + 255) / 256;     // elementwise over [E,256]
  const int gS = (Ee * HD + 255) / 256;     // edge scatter, C=256

  auto gcn_agg = [&](const float* hin, float* agg, const float* dinv,
                     const float* bias, int R, int relu) {
    int g = (R * HD + 255) / 256;
    k_fill<<<g, 256, 0, stream>>>(agg, 0.f, R * HD);
    k_gcn_scatter<<<gS, 256, 0, stream>>>(hin, agg, src, dst, dinv, Ee, HD);
    k_selfbias<<<g, 256, 0, stream>>>(agg, hin, dinv, bias, R, HD, relu);
  };

  // ---- node encoder: relu(gcn c1) -> gcn c2 ----
  k_small_gemm<<<gN, 256, 0, stream>>>(x, F(4), bufN2, Nn, 8, HD);
  gcn_agg(bufN2, bufN1, dinvN, F(5), Nn, 1);
  k_gemm256<<<Nn / 32, 256, 0, stream>>>(bufN1, wp(0), bufN2);
  gcn_agg(bufN2, bufN1, dinvN, F(7), Nn, 0);          // hx = bufN1

  // ---- edge encoder: relu(gcn c1) -> gcn c2 (over E rows, dinvE) ----
  k_small_gemm<<<gE, 256, 0, stream>>>(eattr, F(8), bufE2, Ee, 9, HD);
  gcn_agg(bufE2, bufE1, dinvE, F(9), Ee, 1);
  k_gemm256<<<Ee / 32, 256, 0, stream>>>(bufE1, wp(1), bufE2);
  gcn_agg(bufE2, bufE1, dinvE, F(11), Ee, 0);         // he = bufE1

  // ---- processor: 9 residual message-passing layers ----
  for (int L = 0; L < 9; ++L) {
    int b = 12 + 12 * L;
    // ea = mlp(he) with edge MLP
    k_mlp<<<Ee / 32, 256, 0, stream>>>(bufE1, wp(3 + 4 * L + 2), F(b + 7),
                                       wp(3 + 4 * L + 3), F(b + 9),
                                       F(b + 10), F(b + 11), nullptr, bufE2);
    // agg[dst] += hx[src] + ea
    k_fill<<<gN, 256, 0, stream>>>(bufN2, 0.f, Nn * HD);
    k_msg_scatter<<<gS, 256, 0, stream>>>(bufN1, bufE2, bufN2, src, dst, Ee);
    // hx = mlp(agg) + hx
    k_mlp<<<Nn / 32, 256, 0, stream>>>(bufN2, wp(3 + 4 * L + 0), F(b + 1),
                                       wp(3 + 4 * L + 1), F(b + 3),
                                       F(b + 4), F(b + 5), bufN1, bufN1);
  }

  // ---- decoder ----
  k_gemm256<<<Nn / 32, 256, 0, stream>>>(bufN1, wp(2), bufN2);
  {
    k_fill<<<gN, 256, 0, stream>>>(bufN3, 0.f, Nn * HD);
    k_gcn_scatter<<<gS, 256, 0, stream>>>(bufN2, bufN3, src, dst, dinvN, Ee, HD);
    k_selfbias<<<gN, 256, 0, stream>>>(bufN3, bufN2, dinvN, F(121), Nn, HD, 1);
  }
  // dec c2: [N,256]@[256,3]
  k_small_gemm<<<(Nn * 3 + 255) / 256, 256, 0, stream>>>(bufN3, F(122), h3, Nn, HD, 3);
  k_fill<<<(Nn * 3 + 255) / 256, 256, 0, stream>>>(agg3, 0.f, Nn * 3);
  k_gcn_scatter<<<(Ee * 3 + 255) / 256, 256, 0, stream>>>(h3, agg3, src, dst, dinvN, Ee, 3);
  k_selfbias<<<(Nn * 3 + 255) / 256, 256, 0, stream>>>(agg3, h3, dinvN, F(123), Nn, 3, 0);
  // final: initial[:, :3] + concat(d, initial) @ Wfc + bfc
  k_final<<<(Nn * 3 + 255) / 256, 256, 0, stream>>>(agg3, istate, F(124), F(125),
                                                    (float*)d_out, Nn);
}
